// LeakyRNN_75093208203474
// MI455X (gfx1250) — matrix operand order repaired and checked by reading, not compiled, not run on hardware
//
#include <hip/hip_runtime.h>

// Leaky low-rank RNN scan for MI455X (gfx1250, wave32).
// Grid: 64 blocks (one batch chain per block). Block: 256 threads (8 waves).
// Per step, per wave: ONE v_wmma_f32_16x16x32_f16 computes both of this
// wave's K-chunks of z = tanh(h) @ Vw^T:
//   A rows 0-7  = Vw ranks for chunk c0=2w, rows 8-15 = ranks for c1=2w+1
//   B col 0     = act chunk c0, col 1 = act chunk c1
// The activation exchange is INTRA-WAVE (wave w's chunks cover exactly the
// n-range owned by wave w's threads), done with 8 ds_bpermute_b32 of packed
// f16 pairs -- no LDS staging, no first barrier. Rank partials live in C
// column 0 (lane 0) and column 1 (lane 17) and are combined across waves
// with raw ds_add_f32 under a constant exec mask 0x20001. A 4-deep rotating
// z accumulator makes a SINGLE barrier per step race-free. Barriers fence
// only DScnt so the 8-step-deep register prefetch of x/noise stays in
// flight across them.

#define BB 64
#define TT 1024
#define NN 512
#define RR 8

typedef __attribute__((ext_vector_type(16))) _Float16 v16h;
typedef __attribute__((ext_vector_type(8)))  float    v8f;

// Workgroup barrier that only fences LDS (DScnt), NOT global loads/stores.
// __syncthreads() would insert s_wait_loadcnt/storecnt 0 and destroy the
// deep global prefetch. All cross-thread traffic in this kernel is LDS.
__device__ __forceinline__ void wg_barrier_lds() {
    asm volatile("s_wait_dscnt 0x0\n\t"
                 "s_barrier_signal -1\n\t"
                 "s_barrier_wait -1" ::: "memory");
}

__global__ __launch_bounds__(256) void leaky_rnn_scan_kernel(
    const float* __restrict__ X,   // (B,T,N)
    const float* __restrict__ NZ,  // (B,T,N)
    const float* __restrict__ Vw,  // (R,N)
    const float* __restrict__ Uw,  // (N,R)
    float* __restrict__ OUT)       // (B,T+1,N)
{
    const float ALPHA = 0.1f;                  // DT/TAU
    const float NS    = 0.07071067811865476f;  // sqrt(2*TAU)*G_REC

    const int b    = blockIdx.x;
    const int tid  = threadIdx.x;
    const int lane = tid & 31;
    const int wave = tid >> 5;

    // Only LDS use: 4 rotating z accumulators (race-free with 1 barrier/step).
    __shared__ alignas(32) float zacc[4][16];
    if (tid < 16) {
        zacc[0][tid] = 0.0f; zacc[1][tid] = 0.0f;
        zacc[2][tid] = 0.0f; zacc[3][tid] = 0.0f;
    }

    // Each thread owns h[n0], h[n0+1], n0 = 2*tid (coalesced float2 traffic).
    const int n0 = tid * 2;

    // Resident Uw rows for the rank->N expansion.
    float u0[RR], u1[RR];
    #pragma unroll
    for (int r = 0; r < RR; ++r) {
        u0[r] = Uw[n0 * RR + r];
        u1[r] = Uw[(n0 + 1) * RR + r];
    }

    // Static A fragment (Vw) packing BOTH of this wave's K-chunks:
    // rows m=0..7 = ranks of chunk c0, rows m=8..15 = ranks of chunk c1.
    // A 16x32, contiguous K (B gather matches): lanes 0-15: M=lane, element s holds K=s;
    // lanes 16-31: M=lane-16, element s holds K=16+s.
    const int c0 = wave * 2, c1 = wave * 2 + 1;  // chunks: n in [32c, 32c+32)
    v16h af;
    {
        const int m    = lane & 15;
        const int r    = m & 7;
        const int cc   = (m < 8) ? c0 : c1;
        const int koff = (lane < 16) ? 0 : 16;
        #pragma unroll
        for (int v = 0; v < 8; ++v) {
            #pragma unroll
            for (int hlf = 0; hlf < 2; ++hlf) {
                const int k = koff + 2 * v + hlf;
                af[2 * v + hlf] = (_Float16)Vw[r * NN + cc * 32 + k];
            }
        }
    }

    float h0 = 0.0f, h1 = 0.0f;

    // Output row t=0 is h0 = zeros.
    {
        float2 z2; z2.x = 0.0f; z2.y = 0.0f;
        reinterpret_cast<float2*>(OUT)[(b * (TT + 1)) * (NN / 2) + tid] = z2;
    }

    const float2* Xp  = reinterpret_cast<const float2*>(X);
    const float2* NZp = reinterpret_cast<const float2*>(NZ);

    // ds_bpermute source-lane bases for the intra-wave B-fragment gather.
    // B col 0 (chunk c0): lane 0 <- lanes 0..7, lane 16 <- lanes 8..15.
    // B col 1 (chunk c1): lane 1 <- lanes 16..23, lane 17 <- lanes 24..31.
    // Other lanes gather arbitrary (finite) values: they feed only unused
    // C columns 2..15, so no masking is required (tanh output is finite).
    int srcbase = 0;
    if      (lane == 16) srcbase = 8;
    else if (lane == 1)  srcbase = 16;
    else if (lane == 17) srcbase = 24;
    int bpidx[8];
    #pragma unroll
    for (int j = 0; j < 8; ++j) bpidx[j] = (srcbase + j) * 4;  // byte index

    // LDS byte offsets of the 4 z accumulators (low 32 bits of a flat LDS
    // pointer are the DS address).
    unsigned zaddr[4];
    #pragma unroll
    for (int q = 0; q < 4; ++q)
        zaddr[q] = (unsigned)(unsigned long long)(uintptr_t)&zacc[q][0];

    wg_barrier_lds();  // initial zacc zeros visible to all waves

    // 8-step-deep register prefetch of x and noise (16 loads in flight).
    float2 xs0 = Xp [(b * TT + 0) * (NN / 2) + tid];
    float2 xs1 = Xp [(b * TT + 1) * (NN / 2) + tid];
    float2 xs2 = Xp [(b * TT + 2) * (NN / 2) + tid];
    float2 xs3 = Xp [(b * TT + 3) * (NN / 2) + tid];
    float2 xs4 = Xp [(b * TT + 4) * (NN / 2) + tid];
    float2 xs5 = Xp [(b * TT + 5) * (NN / 2) + tid];
    float2 xs6 = Xp [(b * TT + 6) * (NN / 2) + tid];
    float2 xs7 = Xp [(b * TT + 7) * (NN / 2) + tid];
    float2 ns0 = NZp[(b * TT + 0) * (NN / 2) + tid];
    float2 ns1 = NZp[(b * TT + 1) * (NN / 2) + tid];
    float2 ns2 = NZp[(b * TT + 2) * (NN / 2) + tid];
    float2 ns3 = NZp[(b * TT + 3) * (NN / 2) + tid];
    float2 ns4 = NZp[(b * TT + 4) * (NN / 2) + tid];
    float2 ns5 = NZp[(b * TT + 5) * (NN / 2) + tid];
    float2 ns6 = NZp[(b * TT + 6) * (NN / 2) + tid];
    float2 ns7 = NZp[(b * TT + 7) * (NN / 2) + tid];

    auto step = [&](int t, float2& xc, float2& nc) {
        const int p = t & 3;  // constant after unrolling
        const unsigned zacc_cur = zaddr[p];

        // Native gfx1250 TRANS tanh; trailing V_NOPs cover the TRANS
        // result-use hazard.
        float a0, a1;
        asm volatile("v_tanh_f32 %0, %2\n\t"
                     "v_tanh_f32 %1, %3\n\t"
                     "v_nop\n\t"
                     "v_nop"
                     : "=&v"(a0), "=&v"(a1)
                     : "v"(h0), "v"(h1));

        // Pack two f16 activations into one dword.
        union { _Float16 h2[2]; int i; } pk;
        pk.h2[0] = (_Float16)a0;
        pk.h2[1] = (_Float16)a1;

        // Intra-wave B-fragment gather: 8 pipelined ds_bpermute_b32.
        union { unsigned d[8]; v16h v; } bf;
        #pragma unroll
        for (int j = 0; j < 8; ++j)
            bf.d[j] = (unsigned)__builtin_amdgcn_ds_bpermute(bpidx[j], pk.i);

        // One WMMA: both chunks' rank partials (cols 0 and 1 of C).
        v8f c = {};
        c = __builtin_amdgcn_wmma_f32_16x16x32_f16(false, af, false, bf.v,
                                                   (short)0, c, false, false);

        // Zero the +1 rotation buffer (last read at step t-3, two barriers
        // ago -> race-free with a single barrier per step).
        if (tid < 16) zacc[(p + 1) & 3][tid] = 0.0f;

        // Cross-wave combine: ds_add_f32 from lanes 0 and 17 only, selected
        // by a constant exec mask (no per-step compares/branches; the
        // atomic-optimizer never sees these).
        unsigned esave;
        asm volatile(
            "s_mov_b32 %0, exec_lo\n\t"
            "s_mov_b32 exec_lo, 0x20001\n\t"
            "ds_add_f32 %9, %1\n\t"
            "ds_add_f32 %9, %2 offset:4\n\t"
            "ds_add_f32 %9, %3 offset:8\n\t"
            "ds_add_f32 %9, %4 offset:12\n\t"
            "ds_add_f32 %9, %5 offset:16\n\t"
            "ds_add_f32 %9, %6 offset:20\n\t"
            "ds_add_f32 %9, %7 offset:24\n\t"
            "ds_add_f32 %9, %8 offset:28\n\t"
            "s_mov_b32 exec_lo, %0"
            : "=&s"(esave)
            : "v"(c[0]), "v"(c[1]), "v"(c[2]), "v"(c[3]),
              "v"(c[4]), "v"(c[5]), "v"(c[6]), "v"(c[7]), "v"(zacc_cur)
            : "memory");

        wg_barrier_lds();  // the ONLY barrier per step: z complete

        v8f z = *reinterpret_cast<const v8f*>(&zacc[p][0]);

        // rec = z @ Uw^T for this thread's two elements (v_pk_fma_f32).
        float rec0 = 0.0f, rec1 = 0.0f;
        #pragma unroll
        for (int r = 0; r < RR; ++r) {
            rec0 = fmaf(z[r], u0[r], rec0);
            rec1 = fmaf(z[r], u1[r], rec1);
        }

        h0 = (1.0f - ALPHA) * h0 + ALPHA * (rec0 + NS * nc.x + xc.x);
        h1 = (1.0f - ALPHA) * h1 + ALPHA * (rec1 + NS * nc.y + xc.y);

        float2 o; o.x = h0; o.y = h1;
        reinterpret_cast<float2*>(OUT)[(b * (TT + 1) + t + 1) * (NN / 2) + tid] = o;

        // Refill this slot for step t+8; consumed 8 iterations later ->
        // HBM latency fully hidden (loads fly across the LDS-only barriers).
        if (t + 8 < TT) {
            xc = Xp [(b * TT + t + 8) * (NN / 2) + tid];
            nc = NZp[(b * TT + t + 8) * (NN / 2) + tid];
        }
    };

    #pragma unroll 1
    for (int tb = 0; tb < TT; tb += 8) {
        step(tb + 0, xs0, ns0);
        step(tb + 1, xs1, ns1);
        step(tb + 2, xs2, ns2);
        step(tb + 3, xs3, ns3);
        step(tb + 4, xs4, ns4);
        step(tb + 5, xs5, ns5);
        step(tb + 6, xs6, ns6);
        step(tb + 7, xs7, ns7);
    }
}

extern "C" void kernel_launch(void* const* d_in, const int* in_sizes, int n_in,
                              void* d_out, int out_size, void* d_ws, size_t ws_size,
                              hipStream_t stream) {
    (void)in_sizes; (void)n_in; (void)out_size; (void)d_ws; (void)ws_size;
    const float* X  = (const float*)d_in[0];
    const float* NZ = (const float*)d_in[1];
    const float* Vw = (const float*)d_in[2];
    const float* Uw = (const float*)d_in[3];
    float* OUT = (float*)d_out;
    leaky_rnn_scan_kernel<<<BB, 256, 0, stream>>>(X, NZ, Vw, Uw, OUT);
}
